// SynapsePredictor_26577257628190
// MI455X (gfx1250) — matrix operand order repaired and checked by reading, not compiled, not run on hardware
//
#include <hip/hip_runtime.h>
#include <hip/hip_bf16.h>

// ---------------------------------------------------------------------------
// GraphSAGE (2 layers, mean aggregation) + dot-product edge decoder, MI455X.
// Memory-bound (~1.2 GB traffic -> ~50us at 23.3 TB/s); dense projections use
// v_wmma_f32_16x16x32_f16 with one-time f16/transposed weight staging so the
// B fragments are single vector loads (no per-wave cvt churn).
// ---------------------------------------------------------------------------

#define D_FEAT 64

typedef __attribute__((ext_vector_type(16))) _Float16 v16h;
typedef __attribute__((ext_vector_type(8)))  float    v8f;

// ---------------------------------------------------------------------------
// Zero fill (grid-stride)
// ---------------------------------------------------------------------------
__global__ void zero_kernel(float* __restrict__ p, long n) {
    long i = (long)blockIdx.x * blockDim.x + threadIdx.x;
    long stride = (long)gridDim.x * blockDim.x;
    for (; i < n; i += stride) p[i] = 0.0f;
}

// ---------------------------------------------------------------------------
// One-time weight staging: Wt[n][k] = (f16) W[k][n]   (64x64)
// N-major layout makes each WMMA B-fragment 16 contiguous halfs per lane.
// ---------------------------------------------------------------------------
__global__ void wcvt_kernel(const float* __restrict__ W, _Float16* __restrict__ Wt) {
    int i = blockIdx.x * blockDim.x + threadIdx.x;
    if (i < D_FEAT * D_FEAT) {
        int k = i >> 6, n = i & 63;
        Wt[n * D_FEAT + k] = (_Float16)W[k * D_FEAT + n];
    }
}

// ---------------------------------------------------------------------------
// Edge scatter: agg[dst] += x[src], cnt[dst] += 1.   16 threads per edge,
// each thread handles 4 consecutive floats (float4 coalesced gather).
// ---------------------------------------------------------------------------
__global__ void scatter_kernel(const float* __restrict__ x,
                               const int* __restrict__ src,
                               const int* __restrict__ dst,
                               float* __restrict__ agg,
                               float* __restrict__ cnt,
                               int n_edges) {
    int t = blockIdx.x * blockDim.x + threadIdx.x;
    int e = t >> 4;
    int j = t & 15;
    if (e >= n_edges) return;
    int s = src[e];
    int d = dst[e];
    const float4 v = ((const float4*)(x + (size_t)s * D_FEAT))[j];
    float* dp = agg + (size_t)d * D_FEAT + j * 4;
    atomicAdd(dp + 0, v.x);
    atomicAdd(dp + 1, v.y);
    atomicAdd(dp + 2, v.z);
    atomicAdd(dp + 3, v.w);
    if (j == 0) atomicAdd(cnt + d, 1.0f);
}

// ---------------------------------------------------------------------------
// WMMA fragment loaders.
//
// A-matrix (f16, 16x32, MxK), contiguous k order (same as B):
//   lane L: M = L&15, kh = L>>4
//   halfs [0..7]  hold K = kc + kh*16 +     {0..7}   (contiguous)
//   halfs [8..15] hold K = kc + kh*16 + 8 + {0..7}   (contiguous)
// -> two float4 loads per 8-half run, converted+scaled in registers.
// ---------------------------------------------------------------------------
__device__ __forceinline__ v16h load_a_frag(const float* __restrict__ rowptr,
                                            int kc, int kh, float scale) {
    const float4* p0 = (const float4*)(rowptr + kc + kh * 16);       // 32B aligned
    const float4* p1 = (const float4*)(rowptr + kc + kh * 16 + 8);   // 32B aligned
    float4 u0 = p0[0], u1 = p0[1];
    float4 w0 = p1[0], w1 = p1[1];
    v16h a;
    a[0]  = (_Float16)(u0.x * scale);  a[1]  = (_Float16)(u0.y * scale);
    a[2]  = (_Float16)(u0.z * scale);  a[3]  = (_Float16)(u0.w * scale);
    a[4]  = (_Float16)(u1.x * scale);  a[5]  = (_Float16)(u1.y * scale);
    a[6]  = (_Float16)(u1.z * scale);  a[7]  = (_Float16)(u1.w * scale);
    a[8]  = (_Float16)(w0.x * scale);  a[9]  = (_Float16)(w0.y * scale);
    a[10] = (_Float16)(w0.z * scale);  a[11] = (_Float16)(w0.w * scale);
    a[12] = (_Float16)(w1.x * scale);  a[13] = (_Float16)(w1.y * scale);
    a[14] = (_Float16)(w1.z * scale);  a[15] = (_Float16)(w1.w * scale);
    return a;
}

// B-matrix (f16, 32x16, KxN) from N-major staged weights Wt[64][64]:
//   lane L: N = L&15, kh = L>>4; half h holds K = kc + kh*16 + h
//   -> 16 contiguous halfs at Wt[ncol*64 + kc + kh*16]  (32B aligned)
__device__ __forceinline__ v16h load_b_frag(const _Float16* __restrict__ Wt,
                                            int kc, int kh, int ncol) {
    return *(const v16h*)(Wt + ncol * D_FEAT + kc + kh * 16);
}

// ---------------------------------------------------------------------------
// SAGE layer: out[n] = act( mean[n] @ W_l + xin[n] @ W_r + b )
// mean = agg / max(cnt,1).  One wave computes a 16-node x 64-col tile:
// 4 N-tiles of 16 cols, each accumulating 4 chained WMMAs (2 K-chunks x 2
// GEMMs) into a single f32 accumulator seeded with the bias.  Stores take a
// wave-uniform fast path for full tiles (no per-lane exec juggling).
// ---------------------------------------------------------------------------
template <bool RELU>
__global__ void sage_layer_kernel(const float* __restrict__ agg,
                                  const float* __restrict__ cnt,
                                  const float* __restrict__ xin,
                                  const _Float16* __restrict__ Wtl,
                                  const _Float16* __restrict__ Wtr,
                                  const float* __restrict__ bias,
                                  float* __restrict__ out,
                                  int n_nodes, int n_tiles) {
    int wave = (blockIdx.x * blockDim.x + threadIdx.x) >> 5;
    int lane = threadIdx.x & 31;
    if (wave >= n_tiles) return;              // wave-uniform: EXEC stays all-1s

    int m  = lane & 15;
    int kh = lane >> 4;
    int row  = wave * 16 + m;
    int rowc = row < n_nodes ? row : n_nodes - 1;    // clamp for partial tile
    const bool full = (wave * 16 + 16) <= n_nodes;   // wave-uniform

    const float* aggrow = agg + (size_t)rowc * D_FEAT;
    const float* xrow   = xin + (size_t)rowc * D_FEAT;
    float invc = 1.0f / fmaxf(cnt[rowc], 1.0f);

    // A fragments: mean (scaled agg) and x, two K-chunks of 32 each.
    v16h am0 = load_a_frag(aggrow, 0,  kh, invc);
    v16h am1 = load_a_frag(aggrow, 32, kh, invc);
    v16h ax0 = load_a_frag(xrow,   0,  kh, 1.0f);
    v16h ax1 = load_a_frag(xrow,   32, kh, 1.0f);

    // Output base for this lane's column within the tile: rows wave*16+8*kh+r
    float* obase = out + ((size_t)(wave * 16 + 8 * kh)) * D_FEAT + m;

#pragma unroll
    for (int nt = 0; nt < 4; ++nt) {
        int ncol = nt * 16 + m;
        float bv = bias[ncol];
        v8f c;
#pragma unroll
        for (int r = 0; r < 8; ++r) c[r] = bv;

        c = __builtin_amdgcn_wmma_f32_16x16x32_f16(
                false, am0, false, load_b_frag(Wtl, 0,  kh, ncol), (short)0, c, false, false);
        c = __builtin_amdgcn_wmma_f32_16x16x32_f16(
                false, am1, false, load_b_frag(Wtl, 32, kh, ncol), (short)0, c, false, false);
        c = __builtin_amdgcn_wmma_f32_16x16x32_f16(
                false, ax0, false, load_b_frag(Wtr, 0,  kh, ncol), (short)0, c, false, false);
        c = __builtin_amdgcn_wmma_f32_16x16x32_f16(
                false, ax1, false, load_b_frag(Wtr, 32, kh, ncol), (short)0, c, false, false);

        // C/D layout: vgpr r -> M = r + 8*kh, N = lane&15
        if (full) {
#pragma unroll
            for (int r = 0; r < 8; ++r) {
                float v = c[r];
                if (RELU) v = fmaxf(v, 0.0f);
                obase[(size_t)r * D_FEAT + nt * 16] = v;
            }
        } else {
#pragma unroll
            for (int r = 0; r < 8; ++r) {
                int orow = wave * 16 + r + 8 * kh;
                if (orow < n_nodes) {
                    float v = c[r];
                    if (RELU) v = fmaxf(v, 0.0f);
                    out[(size_t)orow * D_FEAT + ncol] = v;
                }
            }
        }
    }
}

// ---------------------------------------------------------------------------
// Edge decode: out[e] = dot(z[src_e], z[dst_e]).  16 threads per edge,
// float4 loads, shfl_xor reduction within groups of 16 lanes.
// ---------------------------------------------------------------------------
__global__ void decode_kernel(const float* __restrict__ z,
                              const int* __restrict__ es,
                              const int* __restrict__ ed,
                              float* __restrict__ out,
                              int n_label) {
    int t = blockIdx.x * blockDim.x + threadIdx.x;
    int e = t >> 4;
    int j = t & 15;
    if (e >= n_label) return;
    int a = es[e];
    int b = ed[e];
    float4 va = ((const float4*)(z + (size_t)a * D_FEAT))[j];
    float4 vb = ((const float4*)(z + (size_t)b * D_FEAT))[j];
    float s = va.x * vb.x + va.y * vb.y + va.z * vb.z + va.w * vb.w;
    s += __shfl_xor(s, 8, 16);
    s += __shfl_xor(s, 4, 16);
    s += __shfl_xor(s, 2, 16);
    s += __shfl_xor(s, 1, 16);
    if (j == 0) out[e] = s;
}

// ---------------------------------------------------------------------------
// Launch
// ---------------------------------------------------------------------------
extern "C" void kernel_launch(void* const* d_in, const int* in_sizes, int n_in,
                              void* d_out, int out_size, void* d_ws, size_t ws_size,
                              hipStream_t stream) {
    const int n_nodes = in_sizes[0] / D_FEAT;
    const int n_edges = in_sizes[1] / 2;
    const int n_label = in_sizes[2] / 2;

    const float* x    = (const float*)d_in[0];
    const int*   ei   = (const int*)d_in[1];   // [2, E] flat: src then dst
    const int*   eli  = (const int*)d_in[2];   // [2, L] flat
    const float* W1l  = (const float*)d_in[3];
    const float* W1r  = (const float*)d_in[4];
    const float* b1   = (const float*)d_in[5];
    const float* W2l  = (const float*)d_in[6];
    const float* W2r  = (const float*)d_in[7];
    const float* b2   = (const float*)d_in[8];
    float*       outp = (float*)d_out;

    // workspace layout
    float* agg = (float*)d_ws;                       // [N, 64]
    float* cnt = agg + (size_t)n_nodes * D_FEAT;     // [N]
    float* h   = cnt + n_nodes;                      // [N, 64]
    float* z   = h + (size_t)n_nodes * D_FEAT;       // [N, 64]
    _Float16* wt  = (_Float16*)(z + (size_t)n_nodes * D_FEAT);
    _Float16* wt1l = wt;                             // 4 x [64][64] f16, N-major
    _Float16* wt1r = wt + 1 * D_FEAT * D_FEAT;
    _Float16* wt2l = wt + 2 * D_FEAT * D_FEAT;
    _Float16* wt2r = wt + 3 * D_FEAT * D_FEAT;

    const long nz = (long)n_nodes * D_FEAT + n_nodes;     // agg+cnt contiguous
    const int  n_tiles = (n_nodes + 15) / 16;
    const int  layer_blocks   = (n_tiles + 7) / 8;        // 8 waves / block
    const int  scatter_blocks = (n_edges * 16 + 255) / 256;
    const int  decode_blocks  = (n_label * 16 + 255) / 256;
    const int  wcvt_blocks    = (D_FEAT * D_FEAT + 255) / 256;

    const int* e_src = ei;
    const int* e_dst = ei + n_edges;
    const int* l_src = eli;
    const int* l_dst = eli + n_label;

    // ---- stage weights (f16, transposed) ----
    wcvt_kernel<<<wcvt_blocks, 256, 0, stream>>>(W1l, wt1l);
    wcvt_kernel<<<wcvt_blocks, 256, 0, stream>>>(W1r, wt1r);
    wcvt_kernel<<<wcvt_blocks, 256, 0, stream>>>(W2l, wt2l);
    wcvt_kernel<<<wcvt_blocks, 256, 0, stream>>>(W2r, wt2r);

    // ---- layer 1 ----
    zero_kernel<<<2048, 256, 0, stream>>>(agg, nz);
    scatter_kernel<<<scatter_blocks, 256, 0, stream>>>(x, e_src, e_dst, agg, cnt, n_edges);
    sage_layer_kernel<true><<<layer_blocks, 256, 0, stream>>>(
        agg, cnt, x, wt1l, wt1r, b1, h, n_nodes, n_tiles);

    // ---- layer 2 ----
    zero_kernel<<<2048, 256, 0, stream>>>(agg, nz);
    scatter_kernel<<<scatter_blocks, 256, 0, stream>>>(h, e_src, e_dst, agg, cnt, n_edges);
    sage_layer_kernel<false><<<layer_blocks, 256, 0, stream>>>(
        agg, cnt, h, wt2l, wt2r, b2, z, n_nodes, n_tiles);

    // ---- edge decode ----
    decode_kernel<<<decode_blocks, 256, 0, stream>>>(z, l_src, l_dst, outp, n_label);
}
